// down_22574348108531
// MI455X (gfx1250) — matrix operand order repaired and checked by reading, not compiled, not run on hardware
//
#include <hip/hip_runtime.h>
#include <hip/hip_bf16.h>
#include <stdint.h>

typedef __attribute__((ext_vector_type(16))) __bf16 v16bf;
typedef __attribute__((ext_vector_type(8)))  float  v8f;

static __device__ __forceinline__ unsigned short f32_to_bf16(float f) {
    union { float f; unsigned u; } x; x.f = f;
    unsigned r = x.u + 0x7FFFu + ((x.u >> 16) & 1u);   // round-to-nearest-even
    if ((x.u & 0x7F800000u) == 0x7F800000u) r = x.u;   // Inf/NaN: keep bits
    return (unsigned short)(r >> 16);
}

// ---------------------------------------------------------------- zero helpers
__global__ void k_zero16(unsigned short* __restrict__ p, long long n) {
    long long i = (long long)blockIdx.x * blockDim.x + threadIdx.x;
    if (i < n) p[i] = 0;
}
__global__ void k_zero(float* __restrict__ p, int n) {
    int i = blockIdx.x * blockDim.x + threadIdx.x;
    if (i < n) p[i] = 0.f;
}

// ------------------------------------------------- f32 NCHW -> bf16 padded NHWC
// dst layout: [nb][y+1 of H+2][x+1 of W+2][C]; halo pre-zeroed by k_zero16.
__global__ void k_cvt_nhwc(const float* __restrict__ src,
                           unsigned short* __restrict__ dst,
                           int C, int H, int W, long long total) {
    long long i = (long long)blockIdx.x * blockDim.x + threadIdx.x;
    if (i >= total) return;
    int x = (int)(i % W);  long long t = i / W;
    int y = (int)(t % H);  t /= H;
    int c = (int)(t % C);  int nb = (int)(t / C);
    long long d = (((long long)nb * (H + 2) + y + 1) * (W + 2) + x + 1) * C + c;
    dst[d] = f32_to_bf16(src[i]);
}

// ---------------------------------------------------------------- pack weights
// OIHW f32 -> bf16 in 16x32 A-fragment order used here (contiguous k, matches B):
// Wp[((coT*9 + tap)*nChunk + chunk)*512 + lane*16 + slot]
// packed A 16x32 (this code's order): lanes 0-15 row M=lane hold K{0..15},
// lanes 16-31 row M=lane-16 hold K{16..31}; 2 values per VGPR.
__global__ void k_pack_w(const float* __restrict__ W,
                         unsigned short* __restrict__ Wp,
                         int Cin, int nChunk) {
    int i = blockIdx.x * blockDim.x + threadIdx.x;
    int total = 8 * 9 * nChunk * 512;          // coTiles=8 (Cout=128)
    if (i >= total) return;
    int slot  = i & 15;
    int lane  = (i >> 4) & 31;
    int chunk = (i >> 9) % nChunk;
    int rem   = (i >> 9) / nChunk;
    int tap   = rem % 9;
    int coT   = rem / 9;
    int m  = lane & 15;
    int hi = lane >> 4;
    int v  = slot >> 1, p = slot & 1;
    int kbase = hi ? 16 : 0;
    int k  = kbase + v * 2 + p;
    Wp[i] = f32_to_bf16(W[((size_t)(coT * 16 + m) * Cin + chunk * 32 + k) * 9 + tap]);
}

// ---------------------------------------------------------------- WMMA conv3x3
// Implicit GEMM over padded NHWC bf16 input. Block = 256 threads = 8 waves;
// waves = 8 co tiles sharing one 16-wide spatial tile.
// B tile (3 rows x 18 x-positions x CIN) is staged ONCE per block into LDS via
// async global->LDS DMA (ASYNCcnt), then the unrolled K-loop reads B as aligned
// 32B ds loads and A (packed weights) as aligned 32B global loads.
template <int CIN, int H, int W>
__global__ void __launch_bounds__(256)
k_conv3x3_wmma(const unsigned short* __restrict__ Xu,
               const unsigned short* __restrict__ Wp,
               const float* __restrict__ bias,
               float* __restrict__ Y) {
    constexpr int NCHUNK = CIN / 32;
    constexpr int WP   = W + 2;                 // padded row length
    constexpr int LROW = CIN + 16;              // LDS x-stride (+32B: banks/align)
    constexpr int BPSEG = CIN / 8;              // b128 copies per (row,x) segment

    __shared__ __attribute__((aligned(64))) unsigned short smem[54 * LROW];

    const __bf16* X = (const __bf16*)Xu;
    const int tid  = threadIdx.x;
    const int lane = tid & 31;
    const int coT  = tid >> 5;                  // 8 co tiles of 16

    int sp = blockIdx.x;                        // spatial tile id
    const int wT = sp % (W >> 4);  sp /= (W >> 4);
    const int h  = sp % H;
    const int nb = sp / H;
    const int x0 = wT << 4;

    // ---- async stage B tile into LDS (each segment contiguous in both spaces)
    const __bf16* src00 = X + (((size_t)nb * (H + 2) + h) * WP + x0) * CIN;
    for (int t = tid; t < 54 * BPSEG; t += 256) {
        int j   = t % BPSEG;                    // b128 index within segment
        int seg = t / BPSEG;                    // kh*18 + xi
        int xi  = seg % 18, kh = seg / 18;
        const __bf16* g = src00 + ((size_t)kh * WP + xi) * CIN + j * 8;
        unsigned ldsoff = (unsigned)(size_t)(&smem[seg * LROW + j * 8]);
        asm volatile("global_load_async_to_lds_b128 %0, %1, off"
                     :: "v"(ldsoff), "v"(g) : "memory");
    }
    asm volatile("s_wait_asynccnt 0" ::: "memory");
    __syncthreads();

    // B-fragment mapping used here (contiguous k, same as packed A): lanes 0-15 col N=lane, K=0..15;
    // lanes 16-31 col N=lane-16, K=16..31; K consecutive within the lane.
    const int nn = lane & 15;
    const int kb = (lane & 16) ? 16 : 0;
    const unsigned short* Abase =
        Wp + (size_t)(coT * 9) * NCHUNK * 512 + lane * 16;

    v8f acc = {};
    #pragma unroll
    for (int tap = 0; tap < 9; ++tap) {
        const int kh = tap / 3, kw = tap % 3;
        const int brow = (kh * 18 + nn + kw) * LROW + kb;
        if (tap < 8)   // prefetch next tap's packed weights (global_prefetch_b8)
            __builtin_prefetch(Abase + (tap + 1) * NCHUNK * 512, 0, 3);
        #pragma unroll
        for (int ch = 0; ch < NCHUNK; ++ch) {
            v16bf a = *(const v16bf*)(Abase + (tap * NCHUNK + ch) * 512);
            v16bf b = *(const v16bf*)(&smem[brow + ch * 32]);
            acc = __builtin_amdgcn_wmma_f32_16x16x32_bf16(
                      false, a, false, b, (short)0, acc, false, false);
        }
    }

    // C/D layout: VGPR r, lanes 0-15 -> M=r (N=lane); lanes 16-31 -> M=r+8.
    const int rowHi = (lane & 16) ? 8 : 0;
    const size_t HWs = (size_t)H * W;
    float* yb = Y + ((size_t)nb * 128 + coT * 16) * HWs + (size_t)h * W + x0 + nn;
    #pragma unroll
    for (int r = 0; r < 8; ++r) {
        int row = r + rowHi;
        yb[(size_t)row * HWs] = acc[r] + bias[coT * 16 + row];
    }
}

// ---------------------------------------------------------------- BN statistics
__global__ void k_stats(const float* __restrict__ Xr, float* __restrict__ stats,
                        int C, int H, int W) {
    int c  = blockIdx.x % C;
    int nb = blockIdx.x / C;
    const float* p = Xr + ((size_t)nb * C + c) * H * W;
    float s = 0.f, s2 = 0.f;
    int n = H * W;
    for (int i = threadIdx.x; i < n; i += blockDim.x) {
        float v = p[i]; s += v; s2 += v * v;
    }
    __shared__ float sh[512];
    sh[threadIdx.x] = s; sh[256 + threadIdx.x] = s2;
    __syncthreads();
    for (int off = 128; off > 0; off >>= 1) {
        if ((int)threadIdx.x < off) {
            sh[threadIdx.x]       += sh[threadIdx.x + off];
            sh[256 + threadIdx.x] += sh[256 + threadIdx.x + off];
        }
        __syncthreads();
    }
    if (threadIdx.x == 0) {
        atomicAdd(&stats[c],     sh[0]);
        atomicAdd(&stats[C + c], sh[256]);
    }
}

// --------------------------------------- BN+ReLU -> bf16 padded NHWC (next conv)
__global__ void k_bn_relu_nhwc(const float* __restrict__ Xr,
                               const float* __restrict__ stats,
                               const float* __restrict__ gamma,
                               const float* __restrict__ beta,
                               float invCount, int C, int H, int W,
                               long long total,
                               unsigned short* __restrict__ dst) {
    long long i = (long long)blockIdx.x * blockDim.x + threadIdx.x;
    if (i >= total) return;
    int x = (int)(i % W);  long long t = i / W;
    int y = (int)(t % H);  t /= H;
    int c = (int)(t % C);  int nb = (int)(t / C);
    float mean = stats[c] * invCount;
    float var  = stats[C + c] * invCount - mean * mean;
    float inv  = rsqrtf(var + 1e-5f);
    float v = fmaxf((Xr[i] - mean) * inv * gamma[c] + beta[c], 0.f);
    long long d = (((long long)nb * (H + 2) + y + 1) * (W + 2) + x + 1) * C + c;
    dst[d] = f32_to_bf16(v);
}

// --------------------------------------- BN+ReLU -> f32 NCHW (final outputs)
__global__ void k_bn_relu_f32(const float* __restrict__ Xr,
                              const float* __restrict__ stats,
                              const float* __restrict__ gamma,
                              const float* __restrict__ beta,
                              float invCount, int C, long long HW,
                              long long total, float* __restrict__ outF) {
    long long i = (long long)blockIdx.x * blockDim.x + threadIdx.x;
    if (i >= total) return;
    int c = (int)((i / HW) % C);
    float mean = stats[c] * invCount;
    float var  = stats[C + c] * invCount - mean * mean;
    float inv  = rsqrtf(var + 1e-5f);
    outF[i] = fmaxf((Xr[i] - mean) * inv * gamma[c] + beta[c], 0.f);
}

// ------------------------- Haar DWT: skip f32 NCHW -> bf16 padded NHWC (512 ch)
// dst: [nb][h2+1 of 130][w2+1 of 130][512], channel groups cA|cH|cV|cD.
__global__ void k_dwt_nhwc(const float* __restrict__ S,
                           unsigned short* __restrict__ D) {
    const long long H2 = 128, W2 = 128, C = 128;
    long long total = 8 * C * H2 * W2;
    long long i = (long long)blockIdx.x * blockDim.x + threadIdx.x;
    if (i >= total) return;
    long long w2 = i % W2; long long t = i / W2;
    long long h2 = t % H2; t /= H2;
    long long c  = t % C;  long long nb = t / C;
    const float* p = S + ((nb * C + c) * 256 + 2 * h2) * 256 + 2 * w2;
    float a = p[0], b = p[1], cc = p[256], d = p[257];
    long long base = ((nb * 130 + h2 + 1) * 130 + w2 + 1) * 512 + c;
    D[base +   0] = f32_to_bf16((a + b + cc + d) * 0.5f);
    D[base + 128] = f32_to_bf16((a + b - cc - d) * 0.5f);
    D[base + 256] = f32_to_bf16((a - b + cc - d) * 0.5f);
    D[base + 384] = f32_to_bf16((a - b - cc + d) * 0.5f);
}

// ---------------------------------------------------------------- launch
extern "C" void kernel_launch(void* const* d_in, const int* in_sizes, int n_in,
                              void* d_out, int out_size, void* d_ws, size_t ws_size,
                              hipStream_t stream) {
    const float* inp = (const float*)d_in[0];
    const float* W1  = (const float*)d_in[1];
    const float* b1  = (const float*)d_in[2];
    const float* g1  = (const float*)d_in[3];
    const float* be1 = (const float*)d_in[4];
    const float* W2  = (const float*)d_in[5];
    const float* b2  = (const float*)d_in[6];
    const float* g2  = (const float*)d_in[7];
    const float* be2 = (const float*)d_in[8];
    const float* W3  = (const float*)d_in[9];
    const float* b3  = (const float*)d_in[10];
    const float* g3  = (const float*)d_in[11];
    const float* be3 = (const float*)d_in[12];
    float* out = (float*)d_out;
    char*  ws  = (char*)d_ws;

    // workspace layout (bytes)
    unsigned short* xpad = (unsigned short*)ws;                        // 68,161,536
    float*          raw  = (float*)(ws + (size_t)68161536);           // 268,435,456
    unsigned short* hpad = (unsigned short*)(ws + (size_t)336596992); // 138,444,800
    unsigned short* wp   = (unsigned short*)(ws + (size_t)475041792); // 1,179,648
    float*         stats = (float*)(ws + (size_t)476221440);          // 1 KB

    const long long nXpad = 8LL * 258 * 258 * 64;    // 34,080,768 elems
    const long long nHpad = 8LL * 258 * 258 * 128;   // 68,161,536 elems
    const long long nDpad = 8LL * 130 * 130 * 512;   // 69,222,400 elems
    const long long nIn   = 8LL * 64 * 256 * 256;    // 33,554,432
    const long long tot1  = 8LL * 128 * 256 * 256;   // 67,108,864
    const long long totD  = 8LL * 128 * 128 * 128;   // 16,777,216

    // ---- input -> padded NHWC bf16
    k_zero16<<<(int)((nXpad + 255) / 256), 256, 0, stream>>>(xpad, nXpad);
    k_cvt_nhwc<<<(int)((nIn + 255) / 256), 256, 0, stream>>>(inp, xpad, 64, 256, 256, nIn);

    // ---- layer 1: conv(64->128) + BN + ReLU -> padded NHWC bf16
    k_pack_w<<<(8 * 9 * 2 * 512 + 255) / 256, 256, 0, stream>>>(W1, wp, 64, 2);
    k_conv3x3_wmma<64, 256, 256><<<32768, 256, 0, stream>>>(xpad, wp, b1, raw);
    k_zero<<<1, 256, 0, stream>>>(stats, 256);
    k_stats<<<8 * 128, 256, 0, stream>>>(raw, stats, 128, 256, 256);
    k_zero16<<<(int)((nHpad + 255) / 256), 256, 0, stream>>>(hpad, nHpad);
    k_bn_relu_nhwc<<<(int)((tot1 + 255) / 256), 256, 0, stream>>>(
        raw, stats, g1, be1, 1.f / (8.f * 256 * 256), 128, 256, 256, tot1, hpad);

    // ---- layer 2: conv(128->128) + BN + ReLU -> skip (f32 NCHW, d_out)
    k_pack_w<<<(8 * 9 * 4 * 512 + 255) / 256, 256, 0, stream>>>(W2, wp, 128, 4);
    k_conv3x3_wmma<128, 256, 256><<<32768, 256, 0, stream>>>(hpad, wp, b2, raw);
    k_zero<<<1, 256, 0, stream>>>(stats, 256);
    k_stats<<<8 * 128, 256, 0, stream>>>(raw, stats, 128, 256, 256);
    k_bn_relu_f32<<<(int)((tot1 + 255) / 256), 256, 0, stream>>>(
        raw, stats, g2, be2, 1.f / (8.f * 256 * 256), 128, 256LL * 256, tot1, out);

    // ---- Haar DWT -> padded NHWC bf16 (512 ch), reuse hpad
    k_zero16<<<(int)((nDpad + 255) / 256), 256, 0, stream>>>(hpad, nDpad);
    k_dwt_nhwc<<<(int)((totD + 255) / 256), 256, 0, stream>>>(out, hpad);

    // ---- layer 3: conv(512->128, 128x128) + BN + ReLU -> d_out tail (f32)
    k_pack_w<<<(8 * 9 * 16 * 512 + 255) / 256, 256, 0, stream>>>(W3, wp, 512, 16);
    k_conv3x3_wmma<512, 128, 128><<<8192, 256, 0, stream>>>(hpad, wp, b3, raw);
    k_zero<<<1, 256, 0, stream>>>(stats, 256);
    k_stats<<<8 * 128, 256, 0, stream>>>(raw, stats, 128, 128, 128);
    k_bn_relu_f32<<<(int)((totD + 255) / 256), 256, 0, stream>>>(
        raw, stats, g3, be3, 1.f / (8.f * 128 * 128), 128, 128LL * 128, totD,
        out + tot1);
}
